// sn_depthwise_cc_44006234914834
// MI455X (gfx1250) — matrix operand order repaired and checked by reading, not compiled, not run on hardware
//
#include <hip/hip_runtime.h>
#include <hip/hip_bf16.h>

// ---------------------------------------------------------------------------
// CDNA5 (gfx1250) implementation: siamese ResNet backbone + depthwise xcorr.
// All convolutions run as implicit-GEMM on v_wmma_f32_16x16x32_f16 (wave32).
// Fragments are fetched from LDS as 16B-aligned b128 loads in contiguous k
// order (k = 16*(lane/16) + s) for the 16-bit A(16x32)/B(32x16) operands.
// ---------------------------------------------------------------------------

typedef _Float16 h16;
typedef __attribute__((ext_vector_type(16))) _Float16 v16h;
typedef __attribute__((ext_vector_type(8)))  float    v8f;

#define EPS 1e-5f
#define BATCH 32

static inline int cdiv(int a, int b) { return (a + b - 1) / b; }

// ---------------------------------------------------------------------------
// Fold BN (and optional conv bias) into fp16 weights + f32 bias.
// ---------------------------------------------------------------------------
__global__ __launch_bounds__(256) void fold_bn_kernel(
    const float* __restrict__ w, const float* __restrict__ g,
    const float* __restrict__ bb, const float* __restrict__ mm,
    const float* __restrict__ vv, const float* __restrict__ convb,
    h16* __restrict__ wf, float* __restrict__ bf, int O, int K, int total) {
  int idx = blockIdx.x * blockDim.x + threadIdx.x;
  if (idx >= total) return;
  int o = idx / K;
  int k = idx - o * K;
  float inv = g[o] * rsqrtf(vv[o] + EPS);
  wf[idx] = (h16)(w[idx] * inv);
  if (k == 0) {
    float cb = convb ? convb[o] * inv : 0.0f;
    bf[o] = cb + bb[o] - mm[o] * inv;
  }
}

// conv1 special case: 3 identical input channels -> sum over input channels.
__global__ __launch_bounds__(256) void fold_conv1_kernel(
    const float* __restrict__ w, const float* __restrict__ g,
    const float* __restrict__ bb, const float* __restrict__ mm,
    const float* __restrict__ vv, h16* __restrict__ wf, float* __restrict__ bf) {
  int idx = blockIdx.x * blockDim.x + threadIdx.x;
  if (idx >= 64 * 49) return;
  int o = idx / 49;
  int k = idx - o * 49;
  float inv = g[o] * rsqrtf(vv[o] + EPS);
  float s = w[(o * 3 + 0) * 49 + k] + w[(o * 3 + 1) * 49 + k] + w[(o * 3 + 2) * 49 + k];
  wf[idx] = (h16)(s * inv);
  if (k == 0) bf[o] = bb[o] - mm[o] * inv;
}

__global__ __launch_bounds__(256) void f32_to_f16_kernel(
    const float* __restrict__ src, h16* __restrict__ dst, int n) {
  int i = blockIdx.x * blockDim.x + threadIdx.x;
  if (i < n) dst[i] = (h16)src[i];
}

// ---------------------------------------------------------------------------
// Implicit-GEMM convolution via WMMA.
// GEMM: M = B*OH*OW, N = Cout, K = Cin*KH*KW.
// Workgroup: 256 threads = 8 wave32s laid out 4(M) x 2(N-pairs) -> 64x64 C
// tile; each wave computes two 16x16 tiles (A-frag reused) = 2 WMMA / chunk.
// ---------------------------------------------------------------------------
__global__ __launch_bounds__(256) void conv_wmma_kernel(
    const h16* __restrict__ in, const h16* __restrict__ wgt,
    const float* __restrict__ bias, const h16* __restrict__ res,
    h16* __restrict__ out,
    int B, int Cin, int H, int W, int Cout, int OH, int OW,
    int KH, int KW, int stride, int pad, int relu, int K, int M) {
  // rows padded to 40 halves (80 B): keeps every 16B fragment run aligned.
  __shared__ __align__(16) h16 lsA[64][40];   // [m][k]
  __shared__ __align__(16) h16 lsB[64][40];   // [n][k] (K-contiguous per n)

  const int tid  = threadIdx.x;
  const int wave = tid >> 5;
  const int lane = tid & 31;
  const int grp  = lane >> 4;          // lane group (0: lanes 0-15, 1: 16-31)
  const int l16  = lane & 15;
  const int mwav = wave & 3;           // M sub-tile of this wave
  const int nb   = (wave >> 2) * 32;   // N base (two 16-wide sub-tiles)

  const int m0 = blockIdx.x * 64;
  const int n0 = blockIdx.y * 64;
  const int khw = KH * KW;
  const int ohw = OH * OW;
  const bool vecB = (K & 31) == 0;     // all convs except conv1 (K=49)

  v8f acc0 = {}, acc1 = {};

  for (int k0 = 0; k0 < K; k0 += 32) {
    // --- stage A tile: 64(M) x 32(K), im2col gather, zero-padded -----------
    for (int idx = tid; idx < 64 * 32; idx += 256) {
      int lm = idx >> 5;
      int kk = idx & 31;
      int m = m0 + lm;
      int k = k0 + kk;
      h16 v = (h16)0.0f;
      if (m < M && k < K) {
        int c  = k / khw;
        int r  = k - c * khw;
        int ky = r / KW;
        int kx = r - ky * KW;
        int b  = m / ohw;
        int rr = m - b * ohw;
        int oy = rr / OW;
        int ox = rr - oy * OW;
        int iy = oy * stride - pad + ky;
        int ix = ox * stride - pad + kx;
        if (iy >= 0 && iy < H && ix >= 0 && ix < W)
          v = in[((size_t)(b * Cin + c) * H + iy) * W + ix];
      }
      lsA[lm][kk] = v;
    }
    // --- stage B tile: 64(N) x 32(K) ---------------------------------------
    if (vecB) {
      // one 16B copy per thread; prefetch next K-chunk of this row
      int row = tid >> 2;
      int seg = tid & 3;
      const uint4* src =
          (const uint4*)(wgt + (size_t)(n0 + row) * K + k0) + seg;
      if (k0 + 32 < K) __builtin_prefetch((const void*)(src + 4));
      *(uint4*)&lsB[row][seg * 8] = *src;
    } else {
      for (int idx = tid; idx < 64 * 32; idx += 256) {
        int row = idx >> 5;
        int kk  = idx & 31;
        int k   = k0 + kk;
        lsB[row][kk] = (k < K) ? wgt[(size_t)(n0 + row) * K + k] : (h16)0.0f;
      }
    }
    __syncthreads();

    // --- fragments: A = 2x b128 runs, B = 2x b128 runs (contiguous k = 16g + s) ---
    union V { uint4 q[2]; v16h v; };
    V A, B0, B1;
    const h16* ap = &lsA[mwav * 16 + l16][16 * grp];
    A.q[0] = *(const uint4*)ap;          // K = 16g .. 16g+7
    A.q[1] = *(const uint4*)(ap + 8);    // K = 16g+8 .. +15
    const h16* bp0 = &lsB[nb + l16][16 * grp];
    B0.q[0] = *(const uint4*)bp0;        // K = 16g .. 16g+7
    B0.q[1] = *(const uint4*)(bp0 + 8);  // K = 16g+8 .. +15
    const h16* bp1 = &lsB[nb + 16 + l16][16 * grp];
    B1.q[0] = *(const uint4*)bp1;
    B1.q[1] = *(const uint4*)(bp1 + 8);

    acc0 = __builtin_amdgcn_wmma_f32_16x16x32_f16(
        false, A.v, false, B0.v, (short)0, acc0, false, false);
    acc1 = __builtin_amdgcn_wmma_f32_16x16x32_f16(
        false, A.v, false, B1.v, (short)0, acc1, false, false);
    __syncthreads();
  }

  // --- epilogue: C layout VGPR r -> M = r + 8*grp (within 16-tile) ---------
  const int n1 = n0 + nb + l16;
  const int n2 = n1 + 16;
  const float bn1 = bias[n1];
  const float bn2 = bias[n2];
#pragma unroll
  for (int r = 0; r < 8; ++r) {
    int m = m0 + mwav * 16 + r + 8 * grp;
    if (m >= M) continue;
    int b  = m / ohw;
    int rr = m - b * ohw;
    int oy = rr / OW;
    int ox = rr - oy * OW;
    size_t o1 = ((size_t)(b * Cout + n1) * OH + oy) * OW + ox;
    size_t o2 = ((size_t)(b * Cout + n2) * OH + oy) * OW + ox;
    float v1 = acc0[r] + bn1;
    float v2 = acc1[r] + bn2;
    if (res) {
      v1 += (float)res[o1];
      v2 += (float)res[o2];
    }
    if (relu) {
      v1 = fmaxf(v1, 0.0f);
      v2 = fmaxf(v2, 0.0f);
    }
    out[o1] = (h16)v1;
    out[o2] = (h16)v2;
  }
}

// ---------------------------------------------------------------------------
// MaxPool 3x3 stride 2 pad 1 (fp16 NCHW)
// ---------------------------------------------------------------------------
__global__ __launch_bounds__(256) void maxpool_kernel(
    const h16* __restrict__ in, h16* __restrict__ out,
    int B, int C, int H, int W, int OH, int OW) {
  int idx = blockIdx.x * blockDim.x + threadIdx.x;
  int total = B * C * OH * OW;
  if (idx >= total) return;
  int ox = idx % OW;
  int t  = idx / OW;
  int oy = t % OH;
  int bc = t / OH;
  float mx = -3.4e38f;
  for (int dy = 0; dy < 3; ++dy) {
    int iy = oy * 2 - 1 + dy;
    if (iy < 0 || iy >= H) continue;
    for (int dx = 0; dx < 3; ++dx) {
      int ix = ox * 2 - 1 + dx;
      if (ix < 0 || ix >= W) continue;
      mx = fmaxf(mx, (float)in[((size_t)bc * H + iy) * W + ix]);
    }
  }
  out[idx] = (h16)mx;
}

// ---------------------------------------------------------------------------
// Depthwise cross-correlation, pad 9. One block per (b,c); tiles in LDS.
// ---------------------------------------------------------------------------
__global__ __launch_bounds__(384) void xcorr_kernel(
    const h16* __restrict__ f1, const h16* __restrict__ f2,
    h16* __restrict__ out) {
  __shared__ float s1[361];
  __shared__ float s2[361];
  const size_t base = (size_t)blockIdx.x * 361;
  int tid = threadIdx.x;
  if (tid < 361) {
    s1[tid] = (float)f1[base + tid];
    s2[tid] = (float)f2[base + tid];
  }
  __syncthreads();
  if (tid >= 361) return;
  int y = tid / 19, x = tid % 19;
  float acc = 0.0f;
  for (int i = 0; i < 19; ++i) {
    int yy = y + i - 9;
    if (yy < 0 || yy >= 19) continue;
    for (int j = 0; j < 19; ++j) {
      int xx = x + j - 9;
      if (xx < 0 || xx >= 19) continue;
      acc += s1[yy * 19 + xx] * s2[i * 19 + j];
    }
  }
  out[base + tid] = (h16)fmaxf(acc, 0.0f);
}

// ---------------------------------------------------------------------------
// Fused adaptive-avg-pool(1x1) + FC(256->2). One block per batch sample.
// ---------------------------------------------------------------------------
__global__ __launch_bounds__(256) void poolfc_kernel(
    const h16* __restrict__ x, const float* __restrict__ fcw,
    const float* __restrict__ fcb, float* __restrict__ out) {
  __shared__ float sm[256];
  int b = blockIdx.x, c = threadIdx.x;
  const h16* p = x + ((size_t)b * 256 + c) * 361;
  float s = 0.0f;
  for (int i = 0; i < 361; ++i) s += (float)p[i];
  sm[c] = s * (1.0f / 361.0f);
  __syncthreads();
  if (c < 2) {
    float a = fcb[c];
    for (int j = 0; j < 256; ++j) a += sm[j] * fcw[c * 256 + j];
    out[b * 2 + c] = a;
  }
}

// ---------------------------------------------------------------------------
// Host orchestration
// ---------------------------------------------------------------------------
extern "C" void kernel_launch(void* const* d_in, const int* in_sizes, int n_in,
                              void* d_out, int out_size, void* d_ws, size_t ws_size,
                              hipStream_t stream) {
  (void)in_sizes; (void)n_in; (void)out_size; (void)ws_size;
  const int B = BATCH;

  // --- input indices (jax pytree flatten: dict keys sorted, tuples in order)
  // 0:img1 1:img2
  // 2-5: bn1(g,b,m,v)  6: combo1_b  7-10: combo1_bn  11: combo1_w
  // 12: combo2_b  13-16: combo2_bn  17: combo2_w
  // 18: conv1  19: fc_b  20: fc_w
  // layer1[0]: 21-24 bn1, 25-28 bn2, 29 conv1, 30 conv2 ; layer1[1]: 31..40
  // layer2[0]: 41-44 bn1, 45-48 bn2, 49 c1, 50 c2, 51-54 dbn, 55 dconv
  // layer2[1]: 56-59 bn1, 60-63 bn2, 64 c1, 65 c2
  // layer3[0]: 66-69 bn1, 70-73 bn2, 74 c1, 75 c2, 76-79 dbn, 80 dconv
  // layer3[1]: 81-84 bn1, 85-88 bn2, 89 c1, 90 c2
  auto F = [&](int i) { return (const float*)d_in[i]; };

  // --- scratch bump allocator ----------------------------------------------
  size_t off = 0;
  auto alloc = [&](size_t bytes) -> char* {
    off = (off + 255) & ~(size_t)255;
    char* p = (char*)d_ws + off;
    off += bytes;
    return p;
  };
  auto allocH = [&](size_t n) { return (h16*)alloc(n * sizeof(h16)); };
  auto allocF = [&](size_t n) { return (float*)alloc(n * sizeof(float)); };

  // folded weights (fp16, K-contiguous per output channel) + biases (f32)
  h16* Wc1 = allocH(64 * 49);       float* bc1 = allocF(64);
  h16* W10a = allocH(64 * 576);     float* b10a = allocF(64);
  h16* W10b = allocH(64 * 576);     float* b10b = allocF(64);
  h16* W11a = allocH(64 * 576);     float* b11a = allocF(64);
  h16* W11b = allocH(64 * 576);     float* b11b = allocF(64);
  h16* W20a = allocH(128 * 576);    float* b20a = allocF(128);
  h16* W20b = allocH(128 * 1152);   float* b20b = allocF(128);
  h16* W20d = allocH(128 * 64);     float* b20d = allocF(128);
  h16* W21a = allocH(128 * 1152);   float* b21a = allocF(128);
  h16* W21b = allocH(128 * 1152);   float* b21b = allocF(128);
  h16* W30a = allocH(256 * 1152);   float* b30a = allocF(256);
  h16* W30b = allocH(256 * 2304);   float* b30b = allocF(256);
  h16* W30d = allocH(256 * 128);    float* b30d = allocF(256);
  h16* W31a = allocH(256 * 2304);   float* b31a = allocF(256);
  h16* W31b = allocH(256 * 2304);   float* b31b = allocF(256);
  h16* Wcm1 = allocH(256 * 2304);   float* bcm1 = allocF(256);
  h16* Wcm2 = allocH(256 * 2304);   float* bcm2 = allocF(256);

  // activations
  const size_t nImg  = (size_t)B * 299 * 299;
  const size_t nA1   = (size_t)B * 64 * 150 * 150;   // conv1 out (temp pool)
  const size_t n75   = (size_t)B * 64 * 75 * 75;
  const size_t n38   = (size_t)B * 128 * 38 * 38;
  const size_t n19   = (size_t)B * 256 * 19 * 19;
  h16* himg = allocH(nImg);
  h16* act1 = allocH(nA1);
  h16* X75  = allocH(n75);
  h16* fm1  = allocH(n19);
  h16* fm2  = allocH(n19);
  h16* xc   = allocH(n19);
  h16* c1o  = allocH(n19);
  h16* c2o  = allocH(n19);

  // sub-regions of act1 reused across stages (all fit in 46.08M halves)
  h16* T75a = act1;                 // layer1 temps
  h16* T75b = act1 + n75;
  h16* T38a = act1;                 // layer2 temps
  h16* R38  = act1 + n38;
  h16* Y38  = act1 + 2 * n38;
  h16* Z38  = act1 + 3 * n38;
  h16* T19  = act1 + 4 * n38;      // layer3 temps
  h16* R19  = act1 + 4 * n38 + n19;
  h16* Y19  = act1 + 4 * n38 + 2 * n19;

  // --- weight folding ------------------------------------------------------
  auto fold = [&](int wi, int bni, int convbi, int O, int K, h16* wf, float* bf) {
    int total = O * K;
    fold_bn_kernel<<<cdiv(total, 256), 256, 0, stream>>>(
        F(wi), F(bni), F(bni + 1), F(bni + 2), F(bni + 3),
        convbi >= 0 ? F(convbi) : nullptr, wf, bf, O, K, total);
  };
  fold_conv1_kernel<<<cdiv(64 * 49, 256), 256, 0, stream>>>(
      F(18), F(2), F(3), F(4), F(5), Wc1, bc1);
  fold(29, 21, -1, 64, 576, W10a, b10a);
  fold(30, 25, -1, 64, 576, W10b, b10b);
  fold(39, 31, -1, 64, 576, W11a, b11a);
  fold(40, 35, -1, 64, 576, W11b, b11b);
  fold(49, 41, -1, 128, 576, W20a, b20a);
  fold(50, 45, -1, 128, 1152, W20b, b20b);
  fold(55, 51, -1, 128, 64, W20d, b20d);
  fold(64, 56, -1, 128, 1152, W21a, b21a);
  fold(65, 60, -1, 128, 1152, W21b, b21b);
  fold(74, 66, -1, 256, 1152, W30a, b30a);
  fold(75, 70, -1, 256, 2304, W30b, b30b);
  fold(80, 76, -1, 256, 128, W30d, b30d);
  fold(89, 81, -1, 256, 2304, W31a, b31a);
  fold(90, 85, -1, 256, 2304, W31b, b31b);
  fold(11, 7, 6, 256, 2304, Wcm1, bcm1);
  fold(17, 13, 12, 256, 2304, Wcm2, bcm2);

  // --- conv launcher (C tile 64x64 -> grid.y = Cout/64) --------------------
  auto conv = [&](const h16* in, int Cin, int H, int W, const h16* wgt,
                  const float* bias, int Cout, int KH, int KW, int s, int p,
                  const h16* res, int relu, h16* out) {
    int OH = (H + 2 * p - KH) / s + 1;
    int OW = (W + 2 * p - KW) / s + 1;
    int K = Cin * KH * KW;
    int M = B * OH * OW;
    dim3 grid(cdiv(M, 64), Cout / 64);
    conv_wmma_kernel<<<grid, 256, 0, stream>>>(
        in, wgt, bias, res, out, B, Cin, H, W, Cout, OH, OW, KH, KW, s, p,
        relu, K, M);
  };

  // --- backbone ------------------------------------------------------------
  auto backbone = [&](const float* img, h16* fm) {
    f32_to_f16_kernel<<<cdiv((int)nImg, 256), 256, 0, stream>>>(img, himg, (int)nImg);
    // conv1 7x7 s2 p3 (channel-summed weights, Cin=1) + folded BN + ReLU
    conv(himg, 1, 299, 299, Wc1, bc1, 64, 7, 7, 2, 3, nullptr, 1, act1);
    // maxpool 3x3 s2 p1 -> X75
    maxpool_kernel<<<cdiv((int)n75, 256), 256, 0, stream>>>(
        act1, X75, B, 64, 150, 150, 75, 75);
    // layer1 (64, 75x75), identity residuals
    conv(X75, 64, 75, 75, W10a, b10a, 64, 3, 3, 1, 1, nullptr, 1, T75a);
    conv(T75a, 64, 75, 75, W10b, b10b, 64, 3, 3, 1, 1, X75, 1, T75b);
    conv(T75b, 64, 75, 75, W11a, b11a, 64, 3, 3, 1, 1, nullptr, 1, T75a);
    conv(T75a, 64, 75, 75, W11b, b11b, 64, 3, 3, 1, 1, T75b, 1, X75);
    // layer2 (64->128, 75->38), downsample residual in block0
    conv(X75, 64, 75, 75, W20a, b20a, 128, 3, 3, 2, 1, nullptr, 1, T38a);
    conv(X75, 64, 75, 75, W20d, b20d, 128, 1, 1, 2, 0, nullptr, 0, R38);
    conv(T38a, 128, 38, 38, W20b, b20b, 128, 3, 3, 1, 1, R38, 1, Y38);
    conv(Y38, 128, 38, 38, W21a, b21a, 128, 3, 3, 1, 1, nullptr, 1, T38a);
    conv(T38a, 128, 38, 38, W21b, b21b, 128, 3, 3, 1, 1, Y38, 1, Z38);
    // layer3 (128->256, 38->19)
    conv(Z38, 128, 38, 38, W30a, b30a, 256, 3, 3, 2, 1, nullptr, 1, T19);
    conv(Z38, 128, 38, 38, W30d, b30d, 256, 1, 1, 2, 0, nullptr, 0, R19);
    conv(T19, 256, 19, 19, W30b, b30b, 256, 3, 3, 1, 1, R19, 1, Y19);
    conv(Y19, 256, 19, 19, W31a, b31a, 256, 3, 3, 1, 1, nullptr, 1, T19);
    conv(T19, 256, 19, 19, W31b, b31b, 256, 3, 3, 1, 1, Y19, 1, fm);
  };

  backbone((const float*)d_in[0], fm1);
  backbone((const float*)d_in[1], fm2);

  // --- depthwise cross-correlation + ReLU ----------------------------------
  xcorr_kernel<<<B * 256, 384, 0, stream>>>(fm1, fm2, xc);

  // --- combo convs (3x3, conv-bias + BN folded, ReLU) ----------------------
  conv(xc, 256, 19, 19, Wcm1, bcm1, 256, 3, 3, 1, 1, nullptr, 1, c1o);
  conv(c1o, 256, 19, 19, Wcm2, bcm2, 256, 3, 3, 1, 1, nullptr, 1, c2o);

  // --- avgpool + FC --------------------------------------------------------
  poolfc_kernel<<<B, 256, 0, stream>>>(c2o, F(20), F(19), (float*)d_out);
}
